// Bert_BiLSTM_CRF_44495861187135
// MI455X (gfx1250) — matrix operand order repaired and checked by reading, not compiled, not run on hardware
//
#include <hip/hip_runtime.h>
#include <hip/hip_bf16.h>

// ---------------- types for WMMA ----------------
typedef __attribute__((ext_vector_type(16))) _Float16 v16h;
typedef __attribute__((ext_vector_type(8)))  float    v8f;

union AFrag {
    v16h v;
    uint4 q[2];
    _Float16 h[16];
};

// A-fragment loader: row-major row pointer already offset to k0.
// Lane layout chosen here (same K order as B): half=lane>>4 selects contiguous K range
// {16*half..16*half+15}.
__device__ inline v16h load_a_row(const _Float16* p, int half) {
    AFrag af;
    af.q[0] = *(const uint4*)(p + 16 * half);
    af.q[1] = *(const uint4*)(p + 16 * half + 8);
    return af.v;
}
__device__ inline v16h load_a_row(const float* p, int half) {
    AFrag af;
    const float* c0 = p + 16 * half;
    const float* c1 = p + 16 * half + 8;
#pragma unroll
    for (int j = 0; j < 8; ++j) {
        af.h[j]     = (_Float16)c0[j];
        af.h[8 + j] = (_Float16)c1[j];
    }
    return af.v;
}
// B-fragment loader: p = &W[n][k0 + 16*half], 16 contiguous K values.
__device__ inline v16h load_b_row(const _Float16* p) {
    AFrag bf;
    const uint4* q = (const uint4*)p;
    bf.q[0] = q[0];
    bf.q[1] = q[1];
    return bf.v;
}

// ---------- fast branch-free transcendentals (hardware trans ops) ----------
#define LOG2E 1.4426950408889634f

__device__ inline float fast_sigmoid(float x) {
    // 1 / (1 + exp(-x));  exp(-x) = exp2(-x*log2e).  Saturates gracefully:
    // x << 0 -> exp2 -> +inf -> rcp -> 0;  x >> 0 -> exp2 -> 0 -> rcp(1) = 1.
    const float e = __builtin_amdgcn_exp2f(-x * LOG2E);
    return __builtin_amdgcn_rcpf(1.f + e);
}

#if defined(__has_builtin)
#if __has_builtin(__builtin_amdgcn_tanhf)
#define HAVE_AMDGCN_TANH 1
#endif
#endif

__device__ inline float fast_tanh(float x) {
#ifdef HAVE_AMDGCN_TANH
    return __builtin_amdgcn_tanhf(x);
#else
    // tanh(x) = (exp(2x)-1)/(exp(2x)+1); exp2-based, branch free.
    const float t = __builtin_amdgcn_exp2f(x * (2.f * LOG2E));
    return (t - 1.f) * __builtin_amdgcn_rcpf(t + 1.f);
#endif
}

// ---------------- f32 -> f16 convert ----------------
__global__ void f32_to_f16_kernel(const float* __restrict__ src, _Float16* __restrict__ dst, int n) {
    int i = blockIdx.x * 256 + threadIdx.x;
    if (i < n) dst[i] = (_Float16)src[i];
}

// ---------------- gate projection GEMM ----------------
// Out[bt][g] = sum_k X[bt][k] * W[g][k] + bias[g],  G = 512, both directions.
// grid: (BT/16, 512/(16*8), 2 dirs), block = 256 (8 waves), 1 wave = 1 16x16 tile.
template <typename TIn>
__global__ __launch_bounds__(256) void gates_gemm(
    const TIn* __restrict__ X, int Din,
    const _Float16* __restrict__ W0, const float* __restrict__ Bv0, float* __restrict__ O0,
    const _Float16* __restrict__ W1, const float* __restrict__ Bv1, float* __restrict__ O1) {
    const int lane = threadIdx.x & 31;
    const int wave = threadIdx.x >> 5;
    const int half = lane >> 4;
    const int lx   = lane & 15;

    const _Float16* W = blockIdx.z ? W1 : W0;
    const float* Bv   = blockIdx.z ? Bv1 : Bv0;
    float* O          = blockIdx.z ? O1 : O0;

    const int m0    = blockIdx.x * 16;
    const int ntile = blockIdx.y * 8 + wave;
    const int n     = ntile * 16 + lx;

    const TIn* xrow       = X + (size_t)(m0 + lx) * Din;
    const _Float16* wrow  = W + (size_t)n * Din + 16 * half;

    const float bias = Bv[n];
    v8f c;
#pragma unroll
    for (int r = 0; r < 8; ++r) c[r] = bias;

    for (int k0 = 0; k0 < Din; k0 += 32) {
        v16h a = load_a_row(xrow + k0, half);
        v16h b = load_b_row(wrow + k0);
        c = __builtin_amdgcn_wmma_f32_16x16x32_f16(false, a, false, b, (short)0, c, false, false);
    }

    float* orow = O + (size_t)(m0 + 8 * half) * 512 + n;
#pragma unroll
    for (int r = 0; r < 8; ++r) orow[(size_t)r * 512] = c[r];
}

// ---------------- LSTM recurrence (persistent block per direction) ----------------
// blockIdx.x = dir (0=fwd, 1=bwd). 512 threads = 16 waves.
// wave -> (mtile = wave>>3, h-col tile = wave&7); each wave does all 4 gates.
// h (f16) in LDS; c stays in registers shaped like the C/D fragment.
__global__ __launch_bounds__(512) void lstm_rec(
    const float* __restrict__ xgF, const float* __restrict__ xgB,
    const _Float16* __restrict__ whhF, const _Float16* __restrict__ whhB,
    _Float16* __restrict__ enc) {
    __shared__ _Float16 hbuf[32 * 128];

    const int dir = blockIdx.x;
    const float* xg       = dir ? xgB : xgF;
    const _Float16* whh   = dir ? whhB : whhF;
    const int colOff      = dir ? 128 : 0;

    const int lane  = threadIdx.x & 31;
    const int wave  = threadIdx.x >> 5;
    const int half  = lane >> 4;
    const int lx    = lane & 15;
    const int mtile = wave >> 3;
    const int hn    = (wave & 7) * 16 + lx;   // h unit column 0..127
    const int mrow  = mtile * 16 + lx;        // A-fragment row (batch)

    // zero h(t=-1)
    for (int i = threadIdx.x; i < 32 * 128 / 2; i += 512) ((unsigned int*)hbuf)[i] = 0u;
    __syncthreads();

    v8f cst;
#pragma unroll
    for (int r = 0; r < 8; ++r) cst[r] = 0.f;

    for (int t = 0; t < 1024; ++t) {
        const int tt = dir ? (1023 - t) : t;

        // C init = precomputed input projection (includes bias), per gate
        v8f acc[4];
#pragma unroll
        for (int g = 0; g < 4; ++g) {
            const int ncol = g * 128 + hn;
#pragma unroll
            for (int r = 0; r < 8; ++r) {
                const int b = mtile * 16 + r + 8 * half;
                acc[g][r] = xg[((size_t)b * 1024 + tt) * 512 + ncol];
            }
        }

        // += h(t-1) @ whh^T  (K = 128 -> 4 wmma steps, shared A across gates)
#pragma unroll
        for (int kk = 0; kk < 4; ++kk) {
            const int k0 = kk * 32;
            v16h a = load_a_row(hbuf + mrow * 128 + k0, half);
#pragma unroll
            for (int g = 0; g < 4; ++g) {
                v16h bfr = load_b_row(whh + (size_t)(g * 128 + hn) * 128 + k0 + 16 * half);
                acc[g] = __builtin_amdgcn_wmma_f32_16x16x32_f16(false, a, false, bfr, (short)0,
                                                                acc[g], false, false);
            }
        }

        __syncthreads();  // everyone done reading h(t-1)

#pragma unroll
        for (int r = 0; r < 8; ++r) {
            const int b = mtile * 16 + r + 8 * half;
            const float iv = fast_sigmoid(acc[0][r]);
            const float fv = fast_sigmoid(acc[1][r]);
            const float gv = fast_tanh(acc[2][r]);
            const float ov = fast_sigmoid(acc[3][r]);
            const float cn = fv * cst[r] + iv * gv;
            cst[r] = cn;
            const float hv  = ov * fast_tanh(cn);
            const _Float16 h16 = (_Float16)hv;
            hbuf[b * 128 + hn] = h16;
            enc[((size_t)b * 1024 + tt) * 256 + colOff + hn] = h16;
        }
        __syncthreads();  // h(t) visible before next step's reads
    }
}

// ---------------- emissions: enc[BT,256] @ lin_w[4,256]^T + lin_b ----------------
__global__ __launch_bounds__(256) void emis_kernel(const _Float16* __restrict__ enc,
                                                   const float* __restrict__ lw,
                                                   const float* __restrict__ lb,
                                                   float* __restrict__ out) {
    const int bt = blockIdx.x * 256 + threadIdx.x;
    if (bt >= 32768) return;
    float acc0 = lb[0], acc1 = lb[1], acc2 = lb[2], acc3 = lb[3];
    const _Float16* e = enc + (size_t)bt * 256;
#pragma unroll 4
    for (int j = 0; j < 256; ++j) {
        const float v = (float)e[j];
        acc0 += v * lw[j];
        acc1 += v * lw[256 + j];
        acc2 += v * lw[512 + j];
        acc3 += v * lw[768 + j];
    }
    float* o = out + (size_t)bt * 4;
    o[0] = acc0; o[1] = acc1; o[2] = acc2; o[3] = acc3;
}

// ---------------- CRF NLL (K=4): one lane per batch row ----------------
__global__ void crf_kernel(const float* __restrict__ em, const int* __restrict__ tags,
                           const float* __restrict__ start, const float* __restrict__ endv,
                           const float* __restrict__ trans, float* __restrict__ out) {
    __shared__ float red[32];
    const int b = threadIdx.x;
    float tr[16], st[4], en[4];
#pragma unroll
    for (int i = 0; i < 16; ++i) tr[i] = trans[i];
#pragma unroll
    for (int k = 0; k < 4; ++k) { st[k] = start[k]; en[k] = endv[k]; }

    const float* e  = em + (size_t)b * 1024 * 4;
    const int*   tg = tags + (size_t)b * 1024;

    int tp = tg[0];
    float num = st[tp] + e[tp];
    float al[4];
#pragma unroll
    for (int k = 0; k < 4; ++k) al[k] = st[k] + e[k];

    for (int t = 1; t < 1024; ++t) {
        const float* et = e + t * 4;
        const int tc = tg[t];
        num += et[tc] + tr[tp * 4 + tc];
        tp = tc;
        float nx[4];
#pragma unroll
        for (int j = 0; j < 4; ++j) {
            float m = al[0] + tr[j];
#pragma unroll
            for (int i = 1; i < 4; ++i) m = fmaxf(m, al[i] + tr[i * 4 + j]);
            float s = 0.f;
#pragma unroll
            for (int i = 0; i < 4; ++i) s += expf(al[i] + tr[i * 4 + j] - m);
            nx[j] = m + logf(s) + et[j];
        }
#pragma unroll
        for (int k = 0; k < 4; ++k) al[k] = nx[k];
    }
    num += en[tp];
    float m = al[0] + en[0];
#pragma unroll
    for (int k = 1; k < 4; ++k) m = fmaxf(m, al[k] + en[k]);
    float s = 0.f;
#pragma unroll
    for (int k = 0; k < 4; ++k) s += expf(al[k] + en[k] - m);
    const float logZ = m + logf(s);

    red[b] = num - logZ;
    __syncthreads();
    if (b == 0) {
        float acc = 0.f;
        for (int i = 0; i < 32; ++i) acc += red[i];
        out[0] = -acc / 32.f;
    }
}

// ---------------- host launcher ----------------
extern "C" void kernel_launch(void* const* d_in, const int* in_sizes, int n_in,
                              void* d_out, int out_size, void* d_ws, size_t ws_size,
                              hipStream_t stream) {
    (void)in_sizes; (void)n_in; (void)out_size; (void)ws_size;
    const int BT = 32768;

    const float* embeds = (const float*)d_in[0];
    const int*   tags   = (const int*)d_in[1];
    const float* wih0f  = (const float*)d_in[2];
    const float* whh0f  = (const float*)d_in[3];
    const float* b0f    = (const float*)d_in[4];
    const float* wih0b  = (const float*)d_in[5];
    const float* whh0b  = (const float*)d_in[6];
    const float* b0b    = (const float*)d_in[7];
    const float* wih1f  = (const float*)d_in[8];
    const float* whh1f  = (const float*)d_in[9];
    const float* b1f    = (const float*)d_in[10];
    const float* wih1b  = (const float*)d_in[11];
    const float* whh1b  = (const float*)d_in[12];
    const float* b1b    = (const float*)d_in[13];
    const float* lin_w  = (const float*)d_in[14];
    const float* lin_b  = (const float*)d_in[15];
    const float* c_st   = (const float*)d_in[16];
    const float* c_en   = (const float*)d_in[17];
    const float* c_tr   = (const float*)d_in[18];

    char* ws = (char*)d_ws;
    size_t off = 0;
    auto take = [&](size_t bytes) -> void* {
        void* p = ws + off;
        off = (off + bytes + 255) & ~(size_t)255;
        return p;
    };

    _Float16* enc0 = (_Float16*)take((size_t)BT * 256 * 2);
    _Float16* enc1 = (_Float16*)take((size_t)BT * 256 * 2);
    float*    xgf  = (float*)take((size_t)BT * 512 * 4);
    float*    xgb  = (float*)take((size_t)BT * 512 * 4);
    _Float16* w_ih0f = (_Float16*)take((size_t)512 * 768 * 2);
    _Float16* w_ih0b = (_Float16*)take((size_t)512 * 768 * 2);
    _Float16* w_ih1f = (_Float16*)take((size_t)512 * 256 * 2);
    _Float16* w_ih1b = (_Float16*)take((size_t)512 * 256 * 2);
    _Float16* w_hh0f = (_Float16*)take((size_t)512 * 128 * 2);
    _Float16* w_hh0b = (_Float16*)take((size_t)512 * 128 * 2);
    _Float16* w_hh1f = (_Float16*)take((size_t)512 * 128 * 2);
    _Float16* w_hh1b = (_Float16*)take((size_t)512 * 128 * 2);
    float*    emis   = (float*)take((size_t)BT * 4 * 4);

    auto cvt = [&](const float* s, _Float16* d, int n) {
        f32_to_f16_kernel<<<(n + 255) / 256, 256, 0, stream>>>(s, d, n);
    };
    cvt(wih0f, w_ih0f, 512 * 768);
    cvt(wih0b, w_ih0b, 512 * 768);
    cvt(wih1f, w_ih1f, 512 * 256);
    cvt(wih1b, w_ih1b, 512 * 256);
    cvt(whh0f, w_hh0f, 512 * 128);
    cvt(whh0b, w_hh0b, 512 * 128);
    cvt(whh1f, w_hh1f, 512 * 128);
    cvt(whh1b, w_hh1b, 512 * 128);

    dim3 ggrid(BT / 16, 512 / (16 * 8), 2);

    // layer 0
    gates_gemm<float><<<ggrid, 256, 0, stream>>>(embeds, 768,
                                                 w_ih0f, b0f, xgf,
                                                 w_ih0b, b0b, xgb);
    lstm_rec<<<2, 512, 0, stream>>>(xgf, xgb, w_hh0f, w_hh0b, enc0);

    // layer 1 (reuses xg buffers)
    gates_gemm<_Float16><<<ggrid, 256, 0, stream>>>(enc0, 256,
                                                    w_ih1f, b1f, xgf,
                                                    w_ih1b, b1b, xgb);
    lstm_rec<<<2, 512, 0, stream>>>(xgf, xgb, w_hh1f, w_hh1b, enc1);

    // emissions + CRF
    emis_kernel<<<BT / 256, 256, 0, stream>>>(enc1, lin_w, lin_b, emis);
    crf_kernel<<<1, 32, 0, stream>>>(emis, tags, c_st, c_en, c_tr, (float*)d_out);
}
